// GroupedQueryAttention_66571993088129
// MI455X (gfx1250) — compile-verified, matrix operand order checked by reading, not run on hardware
//
#include <hip/hip_runtime.h>

// ---------------------------------------------------------------------------
// CDNA5 (gfx1250) GQA: bf16 WMMA everywhere, f32 accumulate.
// GEMMs: LDS-staged B tiles via async global->LDS copies (ASYNCcnt),
//        4 WMMAs per A-fragment for 3.2x lower per-lane data traffic.
// ---------------------------------------------------------------------------

typedef __bf16 bf16_t;
typedef __attribute__((ext_vector_type(16))) __bf16 v16bf;
typedef __attribute__((ext_vector_type(8)))  __bf16 v8bf;
typedef __attribute__((ext_vector_type(8)))  float  v8f;

// Types for the async global->LDS copy builtin (param type from the compiler:
// pointer to `int __vector(4)` in the respective address spaces).
typedef int v4i_gcc __attribute__((vector_size(16)));
typedef __attribute__((address_space(1))) v4i_gcc g_v4i;
typedef __attribute__((address_space(3))) v4i_gcc l_v4i;

#if defined(__has_builtin)
#if __has_builtin(__builtin_amdgcn_global_load_async_to_lds_b128)
#define HAVE_ASYNC_COPY 1
#endif
#endif
#ifndef HAVE_ASYNC_COPY
#define HAVE_ASYNC_COPY 0
#endif

__device__ __forceinline__ void wait_async0() {
#if HAVE_ASYNC_COPY
#if __has_builtin(__builtin_amdgcn_s_wait_asynccnt)
    __builtin_amdgcn_s_wait_asynccnt(0);
#else
    __asm__ volatile("s_wait_asynccnt 0" ::: "memory");
#endif
#endif
}

// Build a 16-element bf16 fragment from two contiguous 8-element (16B) chunks.
// CDNA5 16-bit A/B fragment layout: lane (l%16) holds row/col, half (l/16)
// selects K sub-ranges [h*8, h*8+8) and [16+h*8, 16+h*8+8).
__device__ __forceinline__ v16bf make_frag(const bf16_t* p0, const bf16_t* p1) {
    v8bf lo = *(const v8bf*)p0;
    v8bf hi = *(const v8bf*)p1;
    v16bf f;
#pragma unroll
    for (int i = 0; i < 8; ++i) { f[i] = lo[i]; f[8 + i] = hi[i]; }
    return f;
}

__device__ __forceinline__ v8f wmma_bf16(v16bf a, v16bf b, v8f c) {
    return __builtin_amdgcn_wmma_f32_16x16x32_bf16(
        false, a, false, b, (short)0, c, false, false);
}

// ---------------------------------------------------------------------------
// fp32 -> bf16 elementwise convert
// ---------------------------------------------------------------------------
__global__ void cvt_bf16(const float* __restrict__ x, bf16_t* __restrict__ y,
                         size_t n) {
    size_t i = (size_t)blockIdx.x * blockDim.x + threadIdx.x;
    if (i < n) y[i] = (bf16_t)x[i];
}

// fp32 [K,N] -> bf16 transposed [N,K]
__global__ void cvt_transpose_bf16(const float* __restrict__ x,
                                   bf16_t* __restrict__ y, int K, int N) {
    size_t i = (size_t)blockIdx.x * blockDim.x + threadIdx.x;
    size_t total = (size_t)K * N;
    if (i >= total) return;
    size_t n = i / K, k = i % K;
    y[n * (size_t)K + k] = (bf16_t)x[k * (size_t)N + n];
}

// ---------------------------------------------------------------------------
// GEMM: C[M,N] = A[M,K] (bf16 row-major) x Bt[N,K] (bf16, B transposed)
// Block: 256 threads = 8 waves. Block tile 128M x 64N; each wave owns a
// 16M x 64N strip (4 accumulators, one shared A fragment per K-step).
// B tile (64 x 32 bf16) is staged in LDS (double-buffered) with async
// global->LDS copies; row stride padded to 40 elems for conflict-free
// ds_load_b128 column reads.
// MODE 0: f32 row-major   MODE 1: bf16 row-major
// MODE 2: bf16 transposed per (b,g): dst[((b*Gc+g)*64+d)*Ssz + s]
// ---------------------------------------------------------------------------
#define BT_STRIDE 40

template <int MODE>
__global__ __launch_bounds__(256) void gemm_bf16_wmma(
    const bf16_t* __restrict__ A, const bf16_t* __restrict__ Bt,
    void* __restrict__ C, int M, int N, int K, int Ssz) {
    __shared__ __align__(16) bf16_t btile[2][64 * BT_STRIDE];

    const int wave = threadIdx.x >> 5;
    const int lane = threadIdx.x & 31;
    const int row  = lane & 15;
    const int h2   = lane >> 4;
    const int tm = blockIdx.y * 128 + wave * 16;   // wave's M base
    const int n0 = blockIdx.x * 64;                // block's N base

    const bf16_t* ap = A + (size_t)(tm + row) * K;

    // cooperative B-tile stage: thread t copies 16B: row t/4, chunk (t%4)*8
    const int sr = threadIdx.x >> 2;
    const int sc = (threadIdx.x & 3) * 8;
    const bf16_t* bsrc_row = Bt + (size_t)(n0 + sr) * K + sc;

    auto stage = [&](int buf, int k0) {
        const bf16_t* src = bsrc_row + k0;
        bf16_t* dst = &btile[buf][sr * BT_STRIDE + sc];
#if HAVE_ASYNC_COPY
        v4i_gcc* gsrc = (v4i_gcc*)(void*)const_cast<bf16_t*>(src);
        v4i_gcc* ldst = (v4i_gcc*)(void*)dst;
        __builtin_amdgcn_global_load_async_to_lds_b128(
            (g_v4i*)gsrc, (l_v4i*)ldst, 0, 0);
#else
        *(v8bf*)dst = *(const v8bf*)src;
#endif
    };

    v8f acc[4] = {v8f{}, v8f{}, v8f{}, v8f{}};
    const int kiters = K >> 5;

    stage(0, 0);
    for (int i = 0; i < kiters; ++i) {
        const int k0 = i << 5;
        wait_async0();        // this wave's copy of the current tile is done
        __syncthreads();      // every wave's copy is done
        if (i + 1 < kiters) stage((i + 1) & 1, k0 + 32);

        const v16bf af = make_frag(ap + k0 + h2 * 8, ap + k0 + 16 + h2 * 8);
        const bf16_t* lb = &btile[i & 1][0];
#pragma unroll
        for (int nb = 0; nb < 4; ++nb) {
            const bf16_t* br = lb + (nb * 16 + row) * BT_STRIDE;
            const v16bf bf = make_frag(br + h2 * 8, br + 16 + h2 * 8);
            acc[nb] = wmma_bf16(af, bf, acc[nb]);
        }
    }

    // C layout: VGPR r holds (m = r + 8*h2, n = lane%16)
#pragma unroll
    for (int nb = 0; nb < 4; ++nb) {
#pragma unroll
        for (int r = 0; r < 8; ++r) {
            int gm = tm + r + 8 * h2;
            int gn = n0 + nb * 16 + row;
            if constexpr (MODE == 0) {
                ((float*)C)[(size_t)gm * N + gn] = acc[nb][r];
            } else if constexpr (MODE == 1) {
                ((bf16_t*)C)[(size_t)gm * N + gn] = (bf16_t)acc[nb][r];
            } else {
                int b = gm / Ssz, s = gm % Ssz;
                int g = gn >> 6, d = gn & 63;
                int Gc = N >> 6;
                ((bf16_t*)C)[((size_t)(b * Gc + g) * 64 + d) * (size_t)Ssz + s] =
                    (bf16_t)acc[nb][r];
            }
        }
    }
}

// ---------------------------------------------------------------------------
// Causal flash attention, grouped-query. One wave = 16 query rows.
// Q: [B,S,H*64] bf16   K: [B,S,G*64] bf16   Vt: [B,G,64,S] bf16
// O: [B,S,H*64] bf16
// ---------------------------------------------------------------------------
__global__ __launch_bounds__(128) void gqa_attn(
    const bf16_t* __restrict__ Qb, const bf16_t* __restrict__ Kb,
    const bf16_t* __restrict__ Vt, bf16_t* __restrict__ Ob) {
    const int S = 2048, E = 2048, G = 8, GD = 512;

    __shared__ __align__(16) bf16_t pbuf[4][16 * 32];  // per-wave P staging

    const int wave = threadIdx.x >> 5;
    const int lane = threadIdx.x & 31;
    const int row  = lane & 15;
    const int h2   = lane >> 4;
    const int qs = blockIdx.x * 64 + wave * 16;
    const int h = blockIdx.y, b = blockIdx.z;
    const int g = h & (G - 1);               // head h -> group h % G
    bf16_t* pb = pbuf[wave];

    const bf16_t* qbase = Qb + ((size_t)(b * S + qs + row)) * E + h * 64;
    const v16bf aq0 = make_frag(qbase +      h2 * 8, qbase + 16 + h2 * 8);
    const v16bf aq1 = make_frag(qbase + 32 + h2 * 8, qbase + 48 + h2 * 8);

    v8f oacc[4] = {v8f{}, v8f{}, v8f{}, v8f{}};
    float rm[8], rl[8];
#pragma unroll
    for (int r = 0; r < 8; ++r) { rm[r] = -__builtin_inff(); rl[r] = 0.f; }

    const float scale = 0.125f * 1.44269504088896340736f;  // 1/sqrt(64)*log2(e)
    const bf16_t* kbase = Kb + (size_t)b * S * GD + g * 64;
    const bf16_t* vbase = Vt + (size_t)(b * G + g) * 64 * (size_t)S;

    const int kend = qs + 16;                // exclusive key bound (causal)
    for (int kt = 0; kt < kend; kt += 32) {
        const bool full = (kt + 16 < kend);

        v8f s0 = {}, s1 = {};
        {
            const bf16_t* kr = kbase + (size_t)(kt + row) * GD;
            v16bf b0 = make_frag(kr +      h2 * 8, kr + 16 + h2 * 8);
            v16bf b1 = make_frag(kr + 32 + h2 * 8, kr + 48 + h2 * 8);
            s0 = wmma_bf16(aq0, b0, s0);
            s0 = wmma_bf16(aq1, b1, s0);
        }
        if (full) {
            const bf16_t* kr = kbase + (size_t)(kt + 16 + row) * GD;
            v16bf b0 = make_frag(kr +      h2 * 8, kr + 16 + h2 * 8);
            v16bf b1 = make_frag(kr + 32 + h2 * 8, kr + 48 + h2 * 8);
            s1 = wmma_bf16(aq0, b0, s1);
            s1 = wmma_bf16(aq1, b1, s1);
        }

        float t0[8], t1[8], mx[8];
#pragma unroll
        for (int r = 0; r < 8; ++r) {
            const int q = qs + r + 8 * h2;
            t0[r] = (kt + row <= q) ? s0[r] * scale : -__builtin_inff();
            t1[r] = (full && (kt + 16 + row <= q)) ? s1[r] * scale
                                                   : -__builtin_inff();
            mx[r] = fmaxf(t0[r], t1[r]);
        }
#pragma unroll
        for (int r = 0; r < 8; ++r)
#pragma unroll
            for (int m = 1; m < 16; m <<= 1)
                mx[r] = fmaxf(mx[r], __shfl_xor(mx[r], m, 32));

        float p0[8], p1[8], sm[8];
#pragma unroll
        for (int r = 0; r < 8; ++r) {
            const float nm = fmaxf(rm[r], mx[r]);
            const float alpha = exp2f(rm[r] - nm);
            p0[r] = exp2f(t0[r] - nm);
            p1[r] = exp2f(t1[r] - nm);
            sm[r] = p0[r] + p1[r];
            rm[r] = nm;
            rl[r] *= alpha;
#pragma unroll
            for (int nb = 0; nb < 4; ++nb) oacc[nb][r] *= alpha;
        }
#pragma unroll
        for (int r = 0; r < 8; ++r) {
#pragma unroll
            for (int m = 1; m < 16; m <<= 1) sm[r] += __shfl_xor(sm[r], m, 32);
            rl[r] += sm[r];
        }

        // P (C-layout) -> A-layout via per-wave LDS staging
#pragma unroll
        for (int r = 0; r < 8; ++r) {
            const int pr = (r + 8 * h2) * 32;
            pb[pr + row]      = (bf16_t)p0[r];
            pb[pr + 16 + row] = (bf16_t)p1[r];
        }
        __asm__ volatile("s_wait_dscnt 0" ::: "memory");

        const v16bf pf = make_frag(pb + row * 32 + h2 * 8,
                                   pb + row * 32 + 16 + h2 * 8);
        __asm__ volatile("s_wait_dscnt 0" ::: "memory");

#pragma unroll
        for (int nb = 0; nb < 4; ++nb) {
            const int d = nb * 16 + row;
            const bf16_t* vr = vbase + (size_t)d * S + kt;
            v16bf vf;
            if (full) {
                vf = make_frag(vr + h2 * 8, vr + 16 + h2 * 8);
            } else {
                v8bf lo = *(const v8bf*)(vr + h2 * 8);
                vf = v16bf{};
#pragma unroll
                for (int i = 0; i < 8; ++i) vf[i] = lo[i];
            }
            oacc[nb] = wmma_bf16(pf, vf, oacc[nb]);
        }
    }

    float inv[8];
#pragma unroll
    for (int r = 0; r < 8; ++r) inv[r] = 1.0f / rl[r];
#pragma unroll
    for (int r = 0; r < 8; ++r) {
        const size_t orow = (size_t)(b * S + qs + r + 8 * h2) * E + h * 64;
#pragma unroll
        for (int nb = 0; nb < 4; ++nb)
            Ob[orow + nb * 16 + row] = (bf16_t)(oacc[nb][r] * inv[r]);
    }
}

// ---------------------------------------------------------------------------
// Host side
// ---------------------------------------------------------------------------
extern "C" void kernel_launch(void* const* d_in, const int* in_sizes, int n_in,
                              void* d_out, int out_size, void* d_ws,
                              size_t ws_size, hipStream_t stream) {
    const int B = 2, S = 2048, E = 2048, H = 32, G = 8, GD = 512;
    const size_t MS = (size_t)B * S;  // 4096 rows

    const float* X  = (const float*)d_in[0];
    const float* Wq = (const float*)d_in[1];
    const float* Wk = (const float*)d_in[2];
    const float* Wv = (const float*)d_in[3];
    const float* Wo = (const float*)d_in[4];

    char* ws = (char*)d_ws;
    size_t off = 0;
    auto take = [&](size_t bytes) {
        void* p = ws + off;
        off += (bytes + 255) & ~(size_t)255;
        return p;
    };
    bf16_t* Xb  = (bf16_t*)take(MS * E * 2);
    bf16_t* Wqt = (bf16_t*)take((size_t)E * E * 2);
    bf16_t* Wkt = (bf16_t*)take((size_t)E * GD * 2);
    bf16_t* Wvt = (bf16_t*)take((size_t)E * GD * 2);
    bf16_t* Wot = (bf16_t*)take((size_t)E * E * 2);
    bf16_t* Qb  = (bf16_t*)take(MS * E * 2);
    bf16_t* Kb  = (bf16_t*)take(MS * GD * 2);
    bf16_t* Vt  = (bf16_t*)take(MS * GD * 2 + 256);
    bf16_t* Ab  = (bf16_t*)take(MS * E * 2);
    if (off > ws_size) return;

    {
        size_t n = MS * E;
        cvt_bf16<<<dim3((unsigned)((n + 255) / 256)), 256, 0, stream>>>(X, Xb, n);
        size_t nw = (size_t)E * E;
        cvt_transpose_bf16<<<dim3((unsigned)((nw + 255) / 256)), 256, 0, stream>>>(Wq, Wqt, E, E);
        size_t nk = (size_t)E * GD;
        cvt_transpose_bf16<<<dim3((unsigned)((nk + 255) / 256)), 256, 0, stream>>>(Wk, Wkt, E, GD);
        cvt_transpose_bf16<<<dim3((unsigned)((nk + 255) / 256)), 256, 0, stream>>>(Wv, Wvt, E, GD);
        cvt_transpose_bf16<<<dim3((unsigned)((nw + 255) / 256)), 256, 0, stream>>>(Wo, Wot, E, E);
    }

    gemm_bf16_wmma<1><<<dim3(E / 64, (unsigned)(MS / 128)), 256, 0, stream>>>(
        Xb, Wqt, Qb, (int)MS, E, E, S);
    gemm_bf16_wmma<1><<<dim3(GD / 64, (unsigned)(MS / 128)), 256, 0, stream>>>(
        Xb, Wkt, Kb, (int)MS, GD, E, S);
    gemm_bf16_wmma<2><<<dim3(GD / 64, (unsigned)(MS / 128)), 256, 0, stream>>>(
        Xb, Wvt, Vt, (int)MS, GD, E, S);

    gqa_attn<<<dim3(S / 64, H, B), 128, 0, stream>>>(Qb, Kb, Vt, Ab);

    gemm_bf16_wmma<0><<<dim3(E / 64, (unsigned)(MS / 128)), 256, 0, stream>>>(
        Ab, Wot, d_out, (int)MS, E, E, S);
}
